// MultiHeadSelfAttention_51780125720928
// MI455X (gfx1250) — matrix operand order repaired and checked by reading, not compiled, not run on hardware
//
#include <hip/hip_runtime.h>
#include <hip/hip_bf16.h>

typedef __bf16 bf16;
typedef __attribute__((ext_vector_type(16))) __bf16 v16bf;
typedef __attribute__((ext_vector_type(8)))  __bf16 bf16x8;
typedef __attribute__((ext_vector_type(8)))  float   v8f;

__device__ __forceinline__ v8f wmma_bf16(v16bf a, v16bf b, v8f c) {
  // D = A(16x32 bf16) * B(32x16 bf16) + C(16x16 f32)
  return __builtin_amdgcn_wmma_f32_16x16x32_bf16(false, a, false, b, (short)0, c,
                                                 false, false);
}

// ---- A operand (16x32 MxK) from row-major source. Contiguous k order (same as load_b16):
// lane<16 holds row (lane&15), K 0..15 contiguous; lane>=16 same row, K 16..31.
__device__ __forceinline__ v16bf load_a16(const bf16* base, int ld) {
  int lane = threadIdx.x & 31;
  const bf16* p = base + (size_t)(lane & 15) * ld + (lane >> 4) * 16;
  bf16x8 lo = *(const bf16x8*)p;
  bf16x8 hi = *(const bf16x8*)(p + 8);
  v16bf r;
#pragma unroll
  for (int i = 0; i < 8; ++i) { r[i] = lo[i]; r[8 + i] = hi[i]; }
  return r;
}

// A operand with split row mapping m -> {0..7, 16..23} (for S^T key tiles so that
// the P^T B-operand K-index <-> key mapping becomes the identity).
__device__ __forceinline__ v16bf load_a16_rowsplit(const bf16* base, int ld) {
  int lane = threadIdx.x & 31;
  int m = lane & 15;
  int row = (m < 8) ? m : (m + 8);
  const bf16* p = base + (size_t)row * ld + (lane >> 4) * 16;
  bf16x8 lo = *(const bf16x8*)p;
  bf16x8 hi = *(const bf16x8*)(p + 8);
  v16bf r;
#pragma unroll
  for (int i = 0; i < 8; ++i) { r[i] = lo[i]; r[8 + i] = hi[i]; }
  return r;
}

// ---- B operand (32x16 KxN) loaded from row-major B^T ([N x K]):
// lane<16 holds column (lane&15), K=0..15 contiguous; lane>=16 K=16..31.
__device__ __forceinline__ v16bf load_b16(const bf16* baseT, int ld) {
  int lane = threadIdx.x & 31;
  const bf16* p = baseT + (size_t)(lane & 15) * ld + (lane >> 4) * 16;
  bf16x8 lo = *(const bf16x8*)p;
  bf16x8 hi = *(const bf16x8*)(p + 8);
  v16bf r;
#pragma unroll
  for (int i = 0; i < 8; ++i) { r[i] = lo[i]; r[8 + i] = hi[i]; }
  return r;
}

// ---------------------------------------------------------------------------
// GEMM: C[M,N] = A[M,K] * Bt[N,K]^T + bias. Wave tile 64x64 (16 accumulators),
// operands fetched straight from L2 (whole working set fits in 192MB L2).
// ---------------------------------------------------------------------------
__global__ void __launch_bounds__(128) gemm_bias_kernel(
    const bf16* __restrict__ A, int lda,
    const bf16* __restrict__ Bt, int ldb,
    const float* __restrict__ bias,
    void* __restrict__ Cout, int ldc,
    int M, int N, int Kd, int out_f32) {
  int wave = threadIdx.x >> 5;
  int lane = threadIdx.x & 31;
  int wid = blockIdx.x * 4 + wave;
  int tiles_n = N >> 6;
  int tiles_m = M >> 6;
  if (wid >= tiles_m * tiles_n) return;
  int tm = (wid / tiles_n) * 64;
  int tn = (wid % tiles_n) * 64;

  v8f acc[4][4] = {};
  for (int k = 0; k < Kd; k += 32) {
    v16bf a[4], b[4];
#pragma unroll
    for (int i = 0; i < 4; ++i)
      a[i] = load_a16(A + (size_t)(tm + i * 16) * lda + k, lda);
#pragma unroll
    for (int j = 0; j < 4; ++j)
      b[j] = load_b16(Bt + (size_t)(tn + j * 16) * ldb + k, ldb);
#pragma unroll
    for (int i = 0; i < 4; ++i)
#pragma unroll
      for (int j = 0; j < 4; ++j)
        acc[i][j] = wmma_bf16(a[i], b[j], acc[i][j]);
  }

  // C/D layout: lane holds column (lane&15); vgpr r = row r + 8*(lane>>4).
  int cn = lane & 15, hi = lane >> 4;
#pragma unroll
  for (int j = 0; j < 4; ++j) {
    int col = tn + j * 16 + cn;
    float bv = bias[col];
#pragma unroll
    for (int i = 0; i < 4; ++i) {
#pragma unroll
      for (int r = 0; r < 8; ++r) {
        int row = tm + i * 16 + hi * 8 + r;
        float v = acc[i][j][r] + bv;
        if (out_f32) ((float*)Cout)[(size_t)row * ldc + col] = v;
        else         ((bf16*)Cout)[(size_t)row * ldc + col] = (bf16)v;
      }
    }
  }
}

// ---------------------------------------------------------------------------
// Flash attention, one wave = 16 queries of one (b,h). Computes S^T = K*Q^T so
// softmax'd P^T repacks into the P*V B-operand lane-locally (no transpose).
// qkv: [B*N, 2304] bf16 (Q at +0, K at +768, V at +1536, per-head h*64)
// VT : [B*H, 64, 4096] bf16
// out: [B*N, 768] bf16
// ---------------------------------------------------------------------------
__global__ void __launch_bounds__(128) attention_kernel(
    const bf16* __restrict__ qkv, const bf16* __restrict__ VT,
    bf16* __restrict__ attnout) {
  int wave = threadIdx.x >> 5;
  int lane = threadIdx.x & 31;
  int wid = blockIdx.x * 4 + wave;
  if (wid >= 24 * 256) return;
  int qt = wid & 255;
  int bh = wid >> 8;
  int b = bh / 12, h = bh % 12;
  int q0 = qt << 4;

  const bf16* Qbase = qkv + (size_t)(b * 4096 + q0) * 2304 + h * 64;
  const bf16* Kbase = qkv + (size_t)(b * 4096) * 2304 + 768 + h * 64;
  const bf16* Vtb   = VT + (size_t)bh * 64 * 4096;

  // Q^T B-operand tiles (Hd = 64 -> two K-steps), hoisted out of the key loop.
  v16bf bq0 = load_b16(Qbase, 2304);
  v16bf bq1 = load_b16(Qbase + 32, 2304);

  v8f o[4] = {};                 // O^T accumulators: 4 d-tiles x (16x16)
  float m = -1e30f, l = 0.0f;
  const float cs = 0.125f * 1.44269504f;  // SCALE * log2(e), exp in exp2 domain

  for (int kb = 0; kb < 4096; kb += 32) {
    // S^T tiles: st0 covers keys {kb..+7, kb+16..+23}, st1 {kb+8..+15, kb+24..+31}
    const bf16* K0 = Kbase + (size_t)kb * 2304;
    const bf16* K1 = Kbase + (size_t)(kb + 8) * 2304;
    v16bf ka00 = load_a16_rowsplit(K0, 2304);
    v16bf ka01 = load_a16_rowsplit(K0 + 32, 2304);
    v16bf ka10 = load_a16_rowsplit(K1, 2304);
    v16bf ka11 = load_a16_rowsplit(K1 + 32, 2304);
    v8f st0 = {}, st1 = {};
    st0 = wmma_bf16(ka00, bq0, st0);
    st0 = wmma_bf16(ka01, bq1, st0);
    st1 = wmma_bf16(ka10, bq0, st1);
    st1 = wmma_bf16(ka11, bq1, st1);

    // Online softmax in exp2 domain. Each lane owns 16 of the 32 key scores
    // for query (lane&15); the other 16 live in lane^16.
    float s0[8], s1[8];
    float vmax = -1e30f;
#pragma unroll
    for (int r = 0; r < 8; ++r) {
      s0[r] = st0[r] * cs;
      s1[r] = st1[r] * cs;
      vmax = fmaxf(vmax, fmaxf(s0[r], s1[r]));
    }
    vmax = fmaxf(vmax, __shfl_xor(vmax, 16, 32));
    float mn = fmaxf(m, vmax);
    float alpha = exp2f(m - mn);

    v16bf pb;                    // P^T as B operand, built lane-locally
    float sum = 0.0f;
#pragma unroll
    for (int r = 0; r < 8; ++r) {
      float p0 = exp2f(s0[r] - mn);
      float p1 = exp2f(s1[r] - mn);
      sum += p0 + p1;
      pb[r]     = (bf16)p0;
      pb[8 + r] = (bf16)p1;
    }
    sum += __shfl_xor(sum, 16, 32);
    l = l * alpha + sum;
    m = mn;

#pragma unroll
    for (int d = 0; d < 4; ++d)
#pragma unroll
      for (int r = 0; r < 8; ++r) o[d][r] *= alpha;

    // O^T += V^T * P^T   (A = V^T rows, contiguous; K-index == key kb+j)
#pragma unroll
    for (int d = 0; d < 4; ++d) {
      v16bf av = load_a16(Vtb + (size_t)(d * 16) * 4096 + kb, 4096);
      o[d] = wmma_bf16(av, pb, o[d]);
    }
  }

  float inv = 1.0f / l;
  int q = lane & 15, hi = lane >> 4;
#pragma unroll
  for (int d = 0; d < 4; ++d) {
    bf16x8 outv;
#pragma unroll
    for (int r = 0; r < 8; ++r) outv[r] = (bf16)(o[d][r] * inv);
    size_t idx = (size_t)(b * 4096 + q0 + q) * 768 + h * 64 + d * 16 + hi * 8;
    *(bf16x8*)(attnout + idx) = outv;
  }
}

// ---------------------------------------------------------------------------
// Prep kernels
// ---------------------------------------------------------------------------
__global__ void f32_to_bf16_kernel(const float* __restrict__ src,
                                   bf16* __restrict__ dst, int n) {
  int i = blockIdx.x * 256 + threadIdx.x;
  if (i < n) dst[i] = (bf16)src[i];
}

// WT[n*K + k] = W[k*N + n]  (write-coalesced over k)
__global__ void transpose_f32_to_bf16_kernel(const float* __restrict__ W,
                                             bf16* __restrict__ WT, int K, int N) {
  int i = blockIdx.x * 256 + threadIdx.x;
  if (i < K * N) {
    int k = i % K;
    int n = i / K;
    WT[i] = (bf16)W[(size_t)k * N + n];
  }
}

// VT[bh][d][n] = qkv[b*4096+n][1536 + h*64 + d]   (write-coalesced over n)
__global__ void repack_vt_kernel(const bf16* __restrict__ qkv,
                                 bf16* __restrict__ VT) {
  int i = blockIdx.x * 256 + threadIdx.x;  // < 24*64*4096
  int n = i & 4095;
  int t = i >> 12;
  int d = t & 63;
  int bh = t >> 6;
  int b = bh / 12, h = bh % 12;
  VT[i] = qkv[(size_t)(b * 4096 + n) * 2304 + 1536 + h * 64 + d];
}

// ---------------------------------------------------------------------------
extern "C" void kernel_launch(void* const* d_in, const int* in_sizes, int n_in,
                              void* d_out, int out_size, void* d_ws, size_t ws_size,
                              hipStream_t stream) {
  (void)in_sizes; (void)n_in; (void)out_size; (void)ws_size;
  const float* x      = (const float*)d_in[0];
  const float* W_qkv  = (const float*)d_in[1];
  const float* b_qkv  = (const float*)d_in[2];
  const float* W_proj = (const float*)d_in[3];
  const float* b_proj = (const float*)d_in[4];
  float* out = (float*)d_out;

  char* ws = (char*)d_ws;
  bf16* x_bf   = (bf16*)(ws);                 // 8192*768   = 12,582,912 B
  bf16* WqkvT  = (bf16*)(ws + 12582912);      // 2304*768   =  3,538,944 B
  bf16* WprojT = (bf16*)(ws + 16121856);      //  768*768   =  1,179,648 B
  bf16* qkv    = (bf16*)(ws + 17301504);      // 8192*2304  = 37,748,736 B
  bf16* VT     = (bf16*)(ws + 55050240);      // 24*64*4096 = 12,582,912 B
  bf16* attn   = (bf16*)(ws + 67633152);      // 8192*768   = 12,582,912 B
                                              // total 80,216,064 B

  f32_to_bf16_kernel<<<24576, 256, 0, stream>>>(x, x_bf, 8192 * 768);
  transpose_f32_to_bf16_kernel<<<6912, 256, 0, stream>>>(W_qkv, WqkvT, 768, 2304);
  transpose_f32_to_bf16_kernel<<<2304, 256, 0, stream>>>(W_proj, WprojT, 768, 768);

  // QKV: [8192,768] x [768,2304] + b -> bf16 qkv ; 4608 wave-tiles / 4 per block
  gemm_bias_kernel<<<1152, 128, 0, stream>>>(x_bf, 768, WqkvT, 768, b_qkv,
                                             (void*)qkv, 2304, 8192, 2304, 768, 0);

  repack_vt_kernel<<<24576, 256, 0, stream>>>(qkv, VT);

  // Attention: 24 (b,h) x 256 query-tiles = 6144 waves / 4 per block
  attention_kernel<<<1536, 128, 0, stream>>>(qkv, VT, attn);

  // Proj: [8192,768] x [768,768] + b -> f32 out ; 1536 wave-tiles / 4 per block
  gemm_bias_kernel<<<384, 128, 0, stream>>>(attn, 768, WprojT, 768, b_proj,
                                            (void*)out, 768, 8192, 768, 768, 1);
}
